// MSDeformAttnPixelDecoder_74277164417677
// MI455X (gfx1250) — matrix operand order repaired and checked by reading, not compiled, not run on hardware
//
#include <hip/hip_runtime.h>
#include <hip/hip_bf16.h>
#include <math.h>

typedef __attribute__((ext_vector_type(16))) _Float16 v16h;
typedef __attribute__((ext_vector_type(8)))  _Float16 v8h;
typedef __attribute__((ext_vector_type(8)))  float    v8f;

#define BATCH 2
#define SEQ   5376
// GEMM macro-tile: 128x128x32, 256 threads = 8 waves, each wave 32x64 (2x4 WMMA frags)
#define GBM 128
#define GBN 128
#define GBK 32
#define GBKP 48   // LDS row stride in halves (96B -> 16B aligned, reduced bank conflicts)

// ---------------------------------------------------------------------------
// WMMA GEMM:  C[M,N] = A[M,K] * op(B) + bias, f16 inputs, f32 accum.
//   BNK=true : B stored [N,K] row-major (NT, weights / scores)
//   BNK=false: B stored [K,N] row-major (NN, NCHW 1x1 conv / attn*V)
// Batched over blockIdx.z with two-level offsets: off = (z/zdiv)*S1 + (z%zdiv)*S2
// Optional f16 epilogue copy (Ch) so downstream GEMMs consume f16 directly.
// Interior tiles stage global->LDS via GLOBAL_LOAD_ASYNC_TO_LDS_B128 (ASYNCcnt);
// edge tiles fall back to guarded vector loads (K assumed multiple of 8).
// ---------------------------------------------------------------------------
template<bool BNK>
__global__ __launch_bounds__(256) void wmma_gemm(
    const _Float16* __restrict__ A, const _Float16* __restrict__ Bm,
    float* __restrict__ C, _Float16* __restrict__ Ch,
    const float* __restrict__ bias,
    int M, int N, int K, int lda, int ldb, int ldc,
    int zdiv, long long aS1, long long aS2, long long bS1, long long bS2,
    long long cS1, long long cS2, int relu)
{
  __shared__ __align__(16) _Float16 As[GBM * GBKP];
  __shared__ __align__(16) _Float16 Bs[GBN * GBKP];
  int z = blockIdx.z, zo = z / zdiv, zi = z - zo * zdiv;
  A  += zo * aS1 + zi * aS2;
  Bm += zo * bS1 + zi * bS2;
  long long coff = zo * cS1 + zi * cS2;
  C += coff; if (Ch) Ch += coff;
  const int m0 = blockIdx.y * GBM;
  const int n0 = blockIdx.x * GBN;
  const int tid  = threadIdx.x;
  const int lane = tid & 31;
  const int wave = tid >> 5;
  const int wm = (wave >> 1) * 32;   // 4 wave-rows
  const int wn = (wave & 1) * 64;    // 2 wave-cols

  v8f acc[2][4] = {};

  for (int k0 = 0; k0 < K; k0 += GBK) {
    const bool fullK = (k0 + GBK <= K);
    const bool fullA = fullK && (m0 + GBM <= M);
    const bool fullB = BNK && fullK && (n0 + GBN <= N);
    // ---- stage A tile (GBM x GBK halves) ----
#pragma unroll
    for (int it = 0; it < 2; ++it) {
      int c = tid + it * 256;
      int row = c >> 2, kc = (c & 3) << 3;
      int gm = m0 + row, gk = k0 + kc;
      if (fullA) {
        const _Float16* src = A + (long long)gm * lda + gk;
        unsigned dst = (unsigned)(size_t)(&As[row * GBKP + kc]);
        asm volatile("global_load_async_to_lds_b128 %0, %1, off"
                     :: "v"(dst), "v"(src) : "memory");
      } else {
        v8h v = {};
        if (gm < M && gk < K) v = *(const v8h*)(A + (long long)gm * lda + gk);
        *(v8h*)(&As[row * GBKP + kc]) = v;
      }
    }
    // ---- stage B tile (GBN x GBK halves), stored [n][k] in LDS ----
    if (BNK) {
#pragma unroll
      for (int it = 0; it < 2; ++it) {
        int c = tid + it * 256;
        int row = c >> 2, kc = (c & 3) << 3;
        int gn = n0 + row, gk = k0 + kc;
        if (fullB) {
          const _Float16* src = Bm + (long long)gn * ldb + gk;
          unsigned dst = (unsigned)(size_t)(&Bs[row * GBKP + kc]);
          asm volatile("global_load_async_to_lds_b128 %0, %1, off"
                       :: "v"(dst), "v"(src) : "memory");
        } else {
          v8h v = {};
          if (gn < N && gk < K) v = *(const v8h*)(Bm + (long long)gn * ldb + gk);
          *(v8h*)(&Bs[row * GBKP + kc]) = v;
        }
      }
    } else {
#pragma unroll
      for (int it = 0; it < 2; ++it) {
        int c = tid + it * 256;
        int kk = c >> 4;           // 0..31
        int nc = (c & 15) << 3;    // 0..120
        int gk = k0 + kk;
        for (int e = 0; e < 8; ++e) {
          int gn = n0 + nc + e;
          _Float16 hv = (_Float16)0.f;
          if (gk < K && gn < N) hv = Bm[(long long)gk * ldb + gn];
          Bs[(nc + e) * GBKP + kk] = hv;
        }
      }
    }
    if (fullA || fullB)
      asm volatile("s_wait_asynccnt 0" ::: "memory");
    __syncthreads();

    if (k0 + GBK < K) {   // prefetch next A K-tile -> global_prefetch_b8
      int pm = m0 + (tid >> 1); if (pm >= M) pm = M - 1;
      __builtin_prefetch((const void*)(A + (long long)pm * lda + k0 + GBK), 0, 3);
    }

    // ---- fragments per documented CDNA5 16-bit layouts ----
    v16h af[2], bf[4];
    {
      int ar  = lane & 15;
      int kof = (lane >> 4) << 3;   // A: lanes<16 -> k{0..7,16..23}; lanes>=16 -> k{8..15,24..31}
#pragma unroll
      for (int i = 0; i < 2; ++i) {
        const _Float16* p = &As[(wm + i * 16 + ar) * GBKP];
        v8h lo = *(const v8h*)(p + kof);
        v8h hi = *(const v8h*)(p + kof + 16);
        v16h t;
#pragma unroll
        for (int e = 0; e < 8; ++e) { t[e] = lo[e]; t[8 + e] = hi[e]; }
        af[i] = t;
      }
      int kof2 = (lane >> 4) << 3;  // B: lanes<16 -> k{0..7,16..23}; lanes>=16 -> k{8..15,24..31}
#pragma unroll
      for (int j = 0; j < 4; ++j) {
        const _Float16* p = &Bs[(wn + j * 16 + ar) * GBKP];
        v8h lo = *(const v8h*)(p + kof2);
        v8h hi = *(const v8h*)(p + kof2 + 16);
        v16h t;
#pragma unroll
        for (int e = 0; e < 8; ++e) { t[e] = lo[e]; t[8 + e] = hi[e]; }
        bf[j] = t;
      }
    }
#pragma unroll
    for (int i = 0; i < 2; ++i)
#pragma unroll
      for (int j = 0; j < 4; ++j)
        acc[i][j] = __builtin_amdgcn_wmma_f32_16x16x32_f16(
            false, af[i], false, bf[j], (short)0, acc[i][j], false, false);
    __syncthreads();
  }

  // ---- epilogue: C layout VGPR r -> (M = r + 8*(lane>=16), N = lane&15) ----
  int sr = lane >> 4, sc = lane & 15;
#pragma unroll
  for (int i = 0; i < 2; ++i)
#pragma unroll
    for (int j = 0; j < 4; ++j) {
      int gn = n0 + wn + j * 16 + sc;
      float bv = 0.f;
      if (bias && gn < N) bv = bias[gn];
#pragma unroll
      for (int rI = 0; rI < 8; ++rI) {
        int gm = m0 + wm + i * 16 + sr * 8 + rI;
        if (gm < M && gn < N) {
          float v = acc[i][j][rI] + bv;
          if (relu) v = fmaxf(v, 0.f);
          C[(long long)gm * ldc + gn] = v;
          if (Ch) Ch[(long long)gm * ldc + gn] = (_Float16)v;
        }
      }
    }
}

// ---------------------------------------------------------------------------
// Support kernels
// ---------------------------------------------------------------------------
__global__ void k_cvt_h(const float* __restrict__ x, _Float16* __restrict__ y, long long n) {
  long long i = (long long)blockIdx.x * 256 + threadIdx.x;
  if (i < n) y[i] = (_Float16)x[i];
}

__global__ void k_addcvt(const float* __restrict__ a, const float* __restrict__ b,
                         _Float16* __restrict__ y, long long n) {
  long long i = (long long)blockIdx.x * 256 + threadIdx.x;
  if (i < n) y[i] = (_Float16)(a[i] + b[i]);
}

// out[c*R + r] = in[r*C + c]  (32x32 LDS tiles, z-batched)
__global__ __launch_bounds__(256) void k_transpose(const float* __restrict__ in,
    float* __restrict__ out, int R, int Cc, long long inStride, long long outStride) {
  __shared__ float t[32][33];
  in  += (long long)blockIdx.z * inStride;
  out += (long long)blockIdx.z * outStride;
  int r0 = blockIdx.y * 32, c0 = blockIdx.x * 32;
  int tx = threadIdx.x & 31, ty = threadIdx.x >> 5;
  for (int i = ty; i < 32; i += 8) {
    int r = r0 + i, c = c0 + tx;
    if (r < R && c < Cc) t[i][tx] = in[(long long)r * Cc + c];
  }
  __syncthreads();
  for (int i = ty; i < 32; i += 8) {
    int c = c0 + i, r = r0 + tx;
    if (r < R && c < Cc) out[(long long)c * R + r] = t[tx][i];
  }
}

__global__ __launch_bounds__(256) void k_groupnorm(const float* __restrict__ x,
    float* __restrict__ y, _Float16* __restrict__ yh,
    const float* __restrict__ g, const float* __restrict__ b,
    int C, int HW, int groups, long long xbs, long long ybs, long long yhbs, int relu) {
  int grp = blockIdx.x, bb = blockIdx.y;
  x += bb * xbs; y += bb * ybs; if (yh) yh += bb * yhbs;
  int cpg = C / groups;
  long long n = (long long)cpg * HW;
  const float* xg = x + (long long)grp * cpg * HW;
  __shared__ float s1[256], s2[256];
  int t = threadIdx.x;
  float su = 0.f, sq = 0.f;
  for (long long i = t; i < n; i += 256) { float v = xg[i]; su += v; sq += v * v; }
  s1[t] = su; s2[t] = sq; __syncthreads();
  for (int o = 128; o > 0; o >>= 1) { if (t < o) { s1[t] += s1[t + o]; s2[t] += s2[t + o]; } __syncthreads(); }
  float mean = s1[0] / (float)n;
  float inv  = rsqrtf(s2[0] / (float)n - mean * mean + 1e-5f);
  for (long long i = t; i < n; i += 256) {
    int c = grp * cpg + (int)(i / HW);
    long long hw = i % HW;
    float v = (xg[i] - mean) * inv * g[c] + b[c];
    if (relu) v = fmaxf(v, 0.f);
    y[(long long)c * HW + hw] = v;
    if (yh) yh[(long long)c * HW + hw] = (_Float16)v;
  }
}

// y = LN(x + a) over D=256, in-place safe (y may alias x)
__global__ __launch_bounds__(256) void k_lnres(const float* __restrict__ x,
    const float* __restrict__ a, float* __restrict__ y, _Float16* __restrict__ yh,
    const float* __restrict__ g, const float* __restrict__ b) {
  long long row = blockIdx.x;
  int t = threadIdx.x;
  float v = x[row * 256 + t] + a[row * 256 + t];
  __shared__ float s1[256], s2[256];
  s1[t] = v; s2[t] = v * v; __syncthreads();
  for (int o = 128; o > 0; o >>= 1) { if (t < o) { s1[t] += s1[t + o]; s2[t] += s2[t + o]; } __syncthreads(); }
  float m = s1[0] * (1.f / 256.f);
  float inv = rsqrtf(s2[0] * (1.f / 256.f) - m * m + 1e-5f);
  float o = (v - m) * inv * g[t] + b[t];
  y[row * 256 + t] = o;
  if (yh) yh[row * 256 + t] = (_Float16)o;
}

__global__ void k_softmax12(float* __restrict__ a, long long n) {
  long long i = (long long)blockIdx.x * 256 + threadIdx.x;
  if (i >= n) return;
  float* p = a + i * 12;
  float mx = -1e30f;
  for (int k = 0; k < 12; ++k) mx = fmaxf(mx, p[k]);
  float s = 0.f;
  for (int k = 0; k < 12; ++k) { float e = expf(p[k] - mx); p[k] = e; s += e; }
  float inv = 1.f / s;
  for (int k = 0; k < 12; ++k) p[k] *= inv;
}

__global__ void k_softmax_row(const float* __restrict__ x, _Float16* __restrict__ y,
                              int L, float scale, long long rows) {
  long long r = (long long)blockIdx.x * 256 + threadIdx.x;
  if (r >= rows) return;
  const float* p = x + r * L;
  _Float16* q = y + r * L;
  float mx = -1e30f;
  for (int l = 0; l < L; ++l) mx = fmaxf(mx, p[l] * scale);
  float s = 0.f;
  for (int l = 0; l < L; ++l) s += expf(p[l] * scale - mx);
  float inv = 1.f / s;
  for (int l = 0; l < L; ++l) q[l] = (_Float16)(expf(p[l] * scale - mx) * inv);
}

// sine positional encoding + level embedding, token-major [B,S,256]
__global__ void k_pos(float* __restrict__ pos, const float* __restrict__ lev) {
  long long i = (long long)blockIdx.x * 256 + threadIdx.x;
  const long long n = (long long)SEQ * 256;
  if (i >= n) return;
  int c = (int)(i & 255);
  int s = (int)(i >> 8);
  int l, sl, Wd;
  if (s < 256)       { l = 0; sl = s;        Wd = 16; }
  else if (s < 1280) { l = 1; sl = s - 256;  Wd = 32; }
  else               { l = 2; sl = s - 1280; Wd = 64; }
  int yy = sl / Wd, xx = sl % Wd;
  int half = c >> 7, f = c & 127;
  float coord = half ? (float)(xx + 1) : (float)(yy + 1);
  float vv = coord / ((float)Wd + 1e-6f) * 6.28318530718f;
  float dt = powf(10000.f, (float)(f & ~1) / 128.f);
  float p = vv / dt;
  float val = ((f & 1) ? cosf(p) : sinf(p)) + lev[l * 256 + c];
  pos[i] = val;
  pos[n + i] = val;   // batch 1 broadcast
}

// MS-deformable sampling: grid (S, B); 8 waves = 8 heads, 32 lanes = head-dim
__global__ __launch_bounds__(256) void k_deform(const float* __restrict__ val,
    const float* __restrict__ off, const float* __restrict__ aw,
    float* __restrict__ out, _Float16* __restrict__ outh) {
  int s = blockIdx.x, b = blockIdx.y;
  int h = threadIdx.x >> 5, d = threadIdx.x & 31;
  int sl, Wd0;
  if (s < 256)       { sl = s;        Wd0 = 16; }
  else if (s < 1280) { sl = s - 256;  Wd0 = 32; }
  else               { sl = s - 1280; Wd0 = 64; }
  float refx = ((sl % Wd0) + 0.5f) / (float)Wd0;
  float refy = ((sl / Wd0) + 0.5f) / (float)Wd0;
  const int i0s[3]  = {0, 256, 1280};
  const int dims[3] = {16, 32, 64};
  const float* vb = val + (long long)b * SEQ * 256;
  long long bs8 = ((long long)b * SEQ + s) * 8 + h;
  const float* ob = off + bs8 * 24;
  const float* ab = aw + bs8 * 12;
  float acc = 0.f;
#pragma unroll
  for (int l = 0; l < 3; ++l) {
    int Wl = dims[l], base = i0s[l];
#pragma unroll
    for (int p = 0; p < 4; ++p) {
      float w  = ab[l * 4 + p];
      float gx = refx + ob[(l * 4 + p) * 2 + 0] / (float)Wl;
      float gy = refy + ob[(l * 4 + p) * 2 + 1] / (float)Wl;
      float fx = gx * Wl - 0.5f, fy = gy * Wl - 0.5f;
      float x0f = floorf(fx), y0f = floorf(fy);
      float wx = fx - x0f, wy = fy - y0f;
      int x0 = (int)x0f, y0 = (int)y0f;
      float sv = 0.f;
#pragma unroll
      for (int dy = 0; dy < 2; ++dy)
#pragma unroll
        for (int dx = 0; dx < 2; ++dx) {
          int xi = x0 + dx, yi = y0 + dy;
          if (xi >= 0 && xi < Wl && yi >= 0 && yi < Wl) {
            float cw = (dx ? wx : 1.f - wx) * (dy ? wy : 1.f - wy);
            sv += cw * vb[((long long)(base + yi * Wl + xi)) * 256 + h * 32 + d];
          }
        }
      acc += w * sv;
    }
  }
  long long oi = ((long long)b * SEQ + s) * 256 + h * 32 + d;
  out[oi] = acc;
  outh[oi] = (_Float16)acc;
}

__global__ __launch_bounds__(256) void k_meanpool(const float* __restrict__ x,
    float* __restrict__ y, _Float16* __restrict__ yh, int HW, long long xbs) {
  int c = blockIdx.x, b = blockIdx.y;
  const float* p = x + b * xbs + (long long)c * HW;
  __shared__ float sm[256];
  float s = 0.f;
  for (int i = threadIdx.x; i < HW; i += 256) s += p[i];
  sm[threadIdx.x] = s; __syncthreads();
  for (int o = 128; o > 0; o >>= 1) { if (threadIdx.x < o) sm[threadIdx.x] += sm[threadIdx.x + o]; __syncthreads(); }
  if (threadIdx.x == 0) { float m = sm[0] / (float)HW; y[b * 512 + c] = m; yh[b * 512 + c] = (_Float16)m; }
}

__global__ void k_broadcast(const float* __restrict__ v, float* __restrict__ out,
                            long long vbs, long long obs, long long n) {
  long long i = (long long)blockIdx.x * 256 + threadIdx.x;
  if (i >= n) return;
  long long hw = i & 4095;
  int c = (int)((i >> 12) & 511);
  int b = (int)(i >> 21);
  out[b * obs + (long long)c * 4096 + hw] = v[b * vbs + c];
}

__global__ void k_bnrelu(const float* __restrict__ x, float* __restrict__ y,
    const float* __restrict__ g, const float* __restrict__ b2,
    const float* __restrict__ rm, const float* __restrict__ rv,
    int C, int HW, long long xbs, long long ybs, long long n) {
  long long i = (long long)blockIdx.x * 256 + threadIdx.x;
  if (i >= n) return;
  long long hw = i % HW;
  int c = (int)((i / HW) % C);
  int b = (int)(i / ((long long)C * HW));
  float s = g[c] * rsqrtf(rv[c] + 1e-5f);
  float v = x[b * xbs + (long long)c * HW + hw] * s + (b2[c] - rm[c] * s);
  y[b * ybs + (long long)c * HW + hw] = fmaxf(v, 0.f);
}

__global__ void k_upnearest(const float* __restrict__ x, float* __restrict__ y,
                            int C, int Hin, int f, long long xbs, long long ybs, long long n) {
  long long i = (long long)blockIdx.x * 256 + threadIdx.x;
  if (i >= n) return;
  int Ho = Hin * f;
  int ox = (int)(i % Ho);
  int oy = (int)((i / Ho) % Ho);
  int c  = (int)((i / ((long long)Ho * Ho)) % C);
  int b  = (int)(i / ((long long)Ho * Ho * C));
  y[b * ybs + ((long long)c * Ho + oy) * Ho + ox] =
      x[b * xbs + ((long long)c * Hin + oy / f) * Hin + ox / f];
}

__global__ void k_bilinear(const float* __restrict__ x, float* __restrict__ y,
                           int C, int Hin, int Ho, long long xbs, long long ybs, long long n) {
  long long i = (long long)blockIdx.x * 256 + threadIdx.x;
  if (i >= n) return;
  int ox = (int)(i % Ho);
  int oy = (int)((i / Ho) % Ho);
  int c  = (int)((i / ((long long)Ho * Ho)) % C);
  int b  = (int)(i / ((long long)Ho * Ho * C));
  float sy = (oy + 0.5f) * (float)Hin / (float)Ho - 0.5f;
  float sx = (ox + 0.5f) * (float)Hin / (float)Ho - 0.5f;
  float y0f = floorf(sy), x0f = floorf(sx);
  float wy = sy - y0f, wx = sx - x0f;
  int y0 = min(max((int)y0f, 0), Hin - 1), y1 = min(max((int)y0f + 1, 0), Hin - 1);
  int x0 = min(max((int)x0f, 0), Hin - 1), x1 = min(max((int)x0f + 1, 0), Hin - 1);
  const float* p = x + b * xbs + (long long)c * Hin * Hin;
  float r0 = p[y0 * Hin + x0] * (1.f - wx) + p[y0 * Hin + x1] * wx;
  float r1 = p[y1 * Hin + x0] * (1.f - wx) + p[y1 * Hin + x1] * wx;
  y[b * ybs + ((long long)c * Ho + oy) * Ho + ox] = r0 * (1.f - wy) + r1 * wy;
}

// direct r x r conv (stride 1, asymmetric pad r/2-1 / r/2) + BN + ReLU
__global__ void k_convr(const float* __restrict__ x, const float* __restrict__ w,
    const float* __restrict__ bias, const float* __restrict__ g, const float* __restrict__ b2,
    const float* __restrict__ rm, const float* __restrict__ rv,
    float* __restrict__ y, int Co, int r, long long xbs, long long ybs, long long n) {
  long long i = (long long)blockIdx.x * 256 + threadIdx.x;
  if (i >= n) return;
  int ox = (int)(i & 63);
  int oy = (int)((i >> 6) & 63);
  int co = (int)((i >> 12) % Co);
  int b  = (int)(i / ((long long)Co * 4096));
  int p0 = r / 2 - 1;
  float acc = bias[co];
  for (int ci = 0; ci < 512; ++ci) {
    const float* xp = x + b * xbs + (long long)ci * 4096;
    const float* wp = w + ((long long)co * 512 + ci) * r * r;
    for (int ky = 0; ky < r; ++ky) {
      int iy = oy + ky - p0;
      if (iy < 0 || iy >= 64) continue;
      for (int kx = 0; kx < r; ++kx) {
        int ix = ox + kx - p0;
        if (ix < 0 || ix >= 64) continue;
        acc += xp[iy * 64 + ix] * wp[ky * r + kx];
      }
    }
  }
  float s = g[co] * rsqrtf(rv[co] + 1e-5f);
  float v = acc * s + (b2[co] - rm[co] * s);
  y[b * ybs + (long long)co * 4096 + (oy * 64 + ox)] = fmaxf(v, 0.f);
}

__device__ __forceinline__ int padmap(int p, int r) {
  int lp = 4 * r - 4;
  if (p < lp) return p + 4 * r + 4;
  p -= lp;
  if (p < 64) return p;
  return 64 - 8 * r + (p - 64);
}

// query windows: q[(b*64+win)*64+pos][ch] = _c[b][ch][...]
__global__ void k_qgather(const float* __restrict__ c, _Float16* __restrict__ q,
                          long long cbs, long long n) {
  long long i = (long long)blockIdx.x * 256 + threadIdx.x;
  if (i >= n) return;
  int ch  = (int)(i & 511);
  int pos = (int)((i >> 9) & 63);
  int win = (int)((i >> 15) & 63);
  int b   = (int)(i >> 21);
  int py = pos >> 3, px = pos & 7, wy = win >> 3, wx = win & 7;
  q[i] = (_Float16)c[b * cbs + (long long)ch * 4096 + (wy * 8 + py) * 64 + (wx * 8 + px)];
}

// kv windows with circular-style context pads of the reference
__global__ void k_kvgather(const float* __restrict__ ctx, _Float16* __restrict__ kv,
                           int r, int co, long long cbs, long long n) {
  long long i = (long long)blockIdx.x * 256 + threadIdx.x;
  if (i >= n) return;
  int feat = (int)(i & 511);
  int pos  = (int)((i >> 9) & 63);
  int win  = (int)((i >> 15) & 63);
  int b    = (int)(i >> 21);
  int ci = feat % co, rr = feat / co;
  int ry = rr / r, rx = rr % r;
  int py = pos >> 3, px = pos & 7, wy = win >> 3, wx = win & 7;
  int row = padmap(8 * wy + py * r + ry, r);
  int col = padmap(8 * wx + px * r + rx, r);
  kv[i] = (_Float16)ctx[b * cbs + (long long)ci * 4096 + row * 64 + col];
}

__global__ void k_oscatter(const float* __restrict__ o, float* __restrict__ out,
                           long long obs, long long n) {
  long long i = (long long)blockIdx.x * 256 + threadIdx.x;
  if (i >= n) return;
  int hw = (int)(i & 4095);
  int ch = (int)((i >> 12) & 511);
  int b  = (int)(i >> 21);
  int yy = hw >> 6, xx = hw & 63;
  int wy = yy >> 3, py = yy & 7, wx = xx >> 3, px = xx & 7;
  out[b * obs + (long long)ch * 4096 + hw] =
      o[(((long long)(b * 64 + wy * 8 + wx)) * 64 + (py * 8 + px)) * 512 + ch];
}

// ---------------------------------------------------------------------------
struct Arena {
  char* base; size_t off;
  void* alloc(size_t bytes) { void* p = base + off; off += (bytes + 255) & ~(size_t)255; return p; }
};
static inline int cdiv(long long a, int b) { return (int)((a + b - 1) / b); }

extern "C" void kernel_launch(void* const* d_in, const int* in_sizes, int n_in,
                              void* d_out, int out_size, void* d_ws, size_t ws_size,
                              hipStream_t stream) {
  (void)in_sizes; (void)n_in; (void)out_size; (void)ws_size;
  Arena ar; ar.base = (char*)d_ws; ar.off = 0;
  auto F = [&](int i) { return (const float*)d_in[i]; };
  auto cvtw = [&](int idx, long long n) -> _Float16* {
    _Float16* p = (_Float16*)ar.alloc(n * sizeof(_Float16));
    k_cvt_h<<<cdiv(n, 256), 256, 0, stream>>>(F(idx), p, n);
    return p;
  };
  auto auf = [&](long long n) { return (float*)ar.alloc(n * sizeof(float)); };
  auto auh = [&](long long n) { return (_Float16*)ar.alloc(n * sizeof(_Float16)); };

  auto gemmNT = [&](const _Float16* A, const _Float16* B, float* C, _Float16* Ch,
                    const float* bias, int M, int N, int K, int lda, int ldb, int ldc,
                    int nz, long long aS, long long bS, long long cS, int relu,
                    int zdiv, long long aS2, long long bS2, long long cS2) {
    dim3 g(cdiv(N, GBN), cdiv(M, GBM), nz);
    wmma_gemm<true><<<g, 256, 0, stream>>>(A, B, C, Ch, bias, M, N, K, lda, ldb, ldc,
                                           zdiv, aS, aS2, bS, bS2, cS, cS2, relu);
  };
  auto gemmNN = [&](const _Float16* A, const _Float16* B, float* C, _Float16* Ch,
                    const float* bias, int M, int N, int K, int lda, int ldb, int ldc,
                    int nz, long long aS, long long bS, long long cS, int relu,
                    int zdiv, long long aS2, long long bS2, long long cS2) {
    dim3 g(cdiv(N, GBN), cdiv(M, GBM), nz);
    wmma_gemm<false><<<g, 256, 0, stream>>>(A, B, C, Ch, bias, M, N, K, lda, ldb, ldc,
                                            zdiv, aS, aS2, bS, bS2, cS, cS2, relu);
  };
  auto gnorm = [&](const float* x, float* y, _Float16* yh, const float* g, const float* b,
                   int C, int HW, long long xbs, long long ybs, long long yhbs, int relu) {
    k_groupnorm<<<dim3(32, BATCH), 256, 0, stream>>>(x, y, yh, g, b, C, HW, 32, xbs, ybs, yhbs, relu);
  };

  // ---------------- input index map (setup_inputs DFS insertion order) ------
  // 0 res2, 1 res3, 2 res4, 3 res5; input_proj[i]: 4+4i (w,b,gn_g,gn_b);
  // 16 level_embed; enc[l] base 17+16l: ow,ob,aw,ab,vw,vb,pw,pb,n1g,n1b,w1,b1,w2,b2,n2g,n2b;
  // 113 lc4(w,g,b) 116 lc3 119 lc2 122 fuse 125 short 128 pool 131 cat 134 low;
  // 137 lc1.w 138-141 lc1.bn(g,b,rm,rv); ds[j] 142+6j: w,b,g,b,rm,rv; attn[j] 160+4j: in_w,in_b,out_w,out_b
  const int lvlHW[3]   = {256, 1024, 4096};
  const int lvlCin[3]  = {2048, 1024, 512};
  const int lvlIn[3]   = {3, 2, 1};     // res5, res4, res3
  const int lvlTok[3]  = {0, 256, 1280};

  // ---------------- weight f16 conversions ----------------------------------
  _Float16* w_proj[3];
  for (int i = 0; i < 3; ++i) w_proj[i] = cvtw(4 + 4 * i, 256LL * lvlCin[i]);
  struct EncW { _Float16 *ow, *aw, *vw, *pw, *w1, *w2; } encw[6];
  for (int l = 0; l < 6; ++l) {
    int e = 17 + 16 * l;
    encw[l].ow = cvtw(e + 0, 192LL * 256);
    encw[l].aw = cvtw(e + 2,  96LL * 256);
    encw[l].vw = cvtw(e + 4, 256LL * 256);
    encw[l].pw = cvtw(e + 6, 256LL * 256);
    encw[l].w1 = cvtw(e + 10, 1024LL * 256);
    encw[l].w2 = cvtw(e + 12, 256LL * 1024);
  }
  _Float16* w_lc4  = cvtw(113, 512LL * 256);
  _Float16* w_lc3  = cvtw(116, 512LL * 256);
  _Float16* w_lc2  = cvtw(119, 512LL * 256);
  _Float16* w_fuse = cvtw(122, 512LL * 1536);
  _Float16* w_shrt = cvtw(125, 512LL * 512);
  _Float16* w_pool = cvtw(128, 512LL * 512);
  _Float16* w_cat  = cvtw(131, 512LL * 2560);
  _Float16* w_low  = cvtw(134, 256LL * 560);
  _Float16* w_lc1  = cvtw(137, 48LL * 256);
  _Float16 *w_inw[3], *w_outw[3];
  for (int j = 0; j < 3; ++j) {
    w_inw[j]  = cvtw(160 + 4 * j, 1536LL * 512);
    w_outw[j] = cvtw(162 + 4 * j, 512LL * 512);
  }

  // ---------------- activation workspace ------------------------------------
  _Float16* h_lvl[3];
  for (int i = 0; i < 3; ++i) {
    long long n = (long long)BATCH * lvlCin[i] * lvlHW[i];
    h_lvl[i] = auh(n);
    k_cvt_h<<<cdiv(n, 256), 256, 0, stream>>>(F(lvlIn[i]), h_lvl[i], n);
  }
  long long nres2 = (long long)BATCH * 256 * 16384;
  _Float16* h_res2 = auh(nres2);
  k_cvt_h<<<cdiv(nres2, 256), 256, 0, stream>>>(F(0), h_res2, nres2);

  const long long NTOK = (long long)BATCH * SEQ * 256;
  float*    buf_chw = auf((long long)BATCH * 256 * 4096);
  float*    x_tok   = auf(NTOK);
  _Float16* x_h     = auh(NTOK);
  float*    pos     = auf(NTOK);
  _Float16* xp_h    = auh(NTOK);
  float*    val_f   = auf(NTOK);
  float*    offs    = auf((long long)BATCH * SEQ * 192);
  float*    alog    = auf((long long)BATCH * SEQ * 96);
  float*    samp    = auf(NTOK);
  _Float16* samp_h  = auh(NTOK);
  float*    attn_o  = auf(NTOK);
  float*    ffh     = auf((long long)BATCH * SEQ * 1024);
  _Float16* ffh_h   = auh((long long)BATCH * SEQ * 1024);
  float*    ffo     = auf(NTOK);

  // ---------------- input projections + groupnorm + tokenize ----------------
  for (int i = 0; i < 3; ++i) {
    int HW = lvlHW[i], Cin = lvlCin[i];
    gemmNN(w_proj[i], h_lvl[i], buf_chw, nullptr, F(5 + 4 * i),
           256, HW, Cin, Cin, HW, HW,
           BATCH, 0, (long long)Cin * HW, 256LL * HW, 0, 1, 0, 0, 0);
    gnorm(buf_chw, buf_chw, nullptr, F(6 + 4 * i), F(7 + 4 * i),
          256, HW, 256LL * HW, 256LL * HW, 0, 0);
    k_transpose<<<dim3(cdiv(HW, 32), 8, BATCH), 256, 0, stream>>>(
        buf_chw, x_tok + (long long)lvlTok[i] * 256, 256, HW,
        256LL * HW, (long long)SEQ * 256);
  }
  k_pos<<<cdiv((long long)SEQ * 256, 256), 256, 0, stream>>>(pos, F(16));
  k_cvt_h<<<cdiv(NTOK, 256), 256, 0, stream>>>(x_tok, x_h, NTOK);
  k_addcvt<<<cdiv(NTOK, 256), 256, 0, stream>>>(x_tok, pos, xp_h, NTOK);

  // ---------------- 6 encoder layers -----------------------------------------
  for (int l = 0; l < 6; ++l) {
    int e = 17 + 16 * l;
    long long tS = (long long)SEQ * 256;
    gemmNT(x_h, encw[l].vw, val_f, nullptr, F(e + 5), SEQ, 256, 256, 256, 256, 256,
           BATCH, tS, 0, tS, 0, 1, 0, 0, 0);
    gemmNT(xp_h, encw[l].ow, offs, nullptr, F(e + 1), SEQ, 192, 256, 256, 256, 192,
           BATCH, tS, 0, (long long)SEQ * 192, 0, 1, 0, 0, 0);
    gemmNT(xp_h, encw[l].aw, alog, nullptr, F(e + 3), SEQ, 96, 256, 256, 256, 96,
           BATCH, tS, 0, (long long)SEQ * 96, 0, 1, 0, 0, 0);
    k_softmax12<<<cdiv((long long)BATCH * SEQ * 8, 256), 256, 0, stream>>>(alog, (long long)BATCH * SEQ * 8);
    k_deform<<<dim3(SEQ, BATCH), 256, 0, stream>>>(val_f, offs, alog, samp, samp_h);
    gemmNT(samp_h, encw[l].pw, attn_o, nullptr, F(e + 7), SEQ, 256, 256, 256, 256, 256,
           BATCH, tS, 0, tS, 0, 1, 0, 0, 0);
    k_lnres<<<BATCH * SEQ, 256, 0, stream>>>(x_tok, attn_o, x_tok, x_h, F(e + 8), F(e + 9));
    gemmNT(x_h, encw[l].w1, ffh, ffh_h, F(e + 11), SEQ, 1024, 256, 256, 256, 1024,
           BATCH, tS, 0, (long long)SEQ * 1024, 1, 1, 0, 0, 0);
    gemmNT(ffh_h, encw[l].w2, ffo, nullptr, F(e + 13), SEQ, 256, 1024, 1024, 1024, 256,
           BATCH, (long long)SEQ * 1024, 0, tS, 0, 1, 0, 0, 0);
    k_lnres<<<BATCH * SEQ, 256, 0, stream>>>(x_tok, ffo, x_tok, x_h, F(e + 14), F(e + 15));
    k_addcvt<<<cdiv(NTOK, 256), 256, 0, stream>>>(x_tok, pos, xp_h, NTOK);
  }

  // ---------------- split tokens back to CHW feature maps --------------------
  _Float16* cxh[3];
  for (int i = 0; i < 3; ++i) {
    int HW = lvlHW[i];
    long long n = (long long)BATCH * 256 * HW;
    k_transpose<<<dim3(8, cdiv(HW, 32), BATCH), 256, 0, stream>>>(
        x_tok + (long long)lvlTok[i] * 256, buf_chw, HW, 256,
        (long long)SEQ * 256, 256LL * HW);
    cxh[i] = auh(n);
    k_cvt_h<<<cdiv(n, 256), 256, 0, stream>>>(buf_chw, cxh[i], n);
  }

  // ---------------- FPN fuse -------------------------------------------------
  float*    tA    = auf((long long)BATCH * 512 * 4096);
  float*    fuseb = auf((long long)BATCH * 1536 * 4096);
  _Float16* fuseh = auh((long long)BATCH * 1536 * 4096);
  float*    cbuf  = auf((long long)BATCH * 512 * 4096);
  _Float16* c_h   = auh((long long)BATCH * 512 * 4096);

  // lc4 (16x16) -> gn/relu -> up x4 into fuse seg 0
  gemmNN(w_lc4, cxh[0], tA, nullptr, nullptr, 512, 256, 256, 256, 256, 256,
         BATCH, 0, 256LL * 256, 512LL * 256, 0, 1, 0, 0, 0);
  gnorm(tA, tA, nullptr, F(114), F(115), 512, 256, 512LL * 256, 512LL * 256, 0, 1);
  k_upnearest<<<cdiv((long long)BATCH * 512 * 4096, 256), 256, 0, stream>>>(
      tA, fuseb, 512, 16, 4, 512LL * 256, 1536LL * 4096, (long long)BATCH * 512 * 4096);
  // lc3 (32x32) -> up x2 into seg 1
  gemmNN(w_lc3, cxh[1], tA, nullptr, nullptr, 512, 1024, 256, 256, 1024, 1024,
         BATCH, 0, 256LL * 1024, 512LL * 1024, 0, 1, 0, 0, 0);
  gnorm(tA, tA, nullptr, F(117), F(118), 512, 1024, 512LL * 1024, 512LL * 1024, 0, 1);
  k_upnearest<<<cdiv((long long)BATCH * 512 * 4096, 256), 256, 0, stream>>>(
      tA, fuseb + 512LL * 4096, 512, 32, 2, 512LL * 1024, 1536LL * 4096, (long long)BATCH * 512 * 4096);
  // lc2 (64x64) -> gn/relu straight into seg 2
  gemmNN(w_lc2, cxh[2], tA, nullptr, nullptr, 512, 4096, 256, 256, 4096, 4096,
         BATCH, 0, 256LL * 4096, 512LL * 4096, 0, 1, 0, 0, 0);
  gnorm(tA, fuseb + 1024LL * 4096, nullptr, F(120), F(121), 512, 4096,
        512LL * 4096, 1536LL * 4096, 0, 1);
  // fuse conv
  long long nfuse = (long long)BATCH * 1536 * 4096;
  k_cvt_h<<<cdiv(nfuse, 256), 256, 0, stream>>>(fuseb, fuseh, nfuse);
  gemmNN(w_fuse, fuseh, cbuf, nullptr, nullptr, 512, 4096, 1536, 1536, 4096, 4096,
         BATCH, 0, 1536LL * 4096, 512LL * 4096, 0, 1, 0, 0, 0);
  gnorm(cbuf, cbuf, c_h, F(123), F(124), 512, 4096, 512LL * 4096, 512LL * 4096, 512LL * 4096, 1);

  // ---------------- 5-branch concat (short / pooled / 3x window-MHA) ---------
  float* catb = auf((long long)BATCH * 2560 * 4096);
  // short
  gemmNN(w_shrt, c_h, tA, nullptr, nullptr, 512, 4096, 512, 512, 4096, 4096,
         BATCH, 0, 512LL * 4096, 512LL * 4096, 0, 1, 0, 0, 0);
  gnorm(tA, catb, nullptr, F(126), F(127), 512, 4096, 512LL * 4096, 2560LL * 4096, 0, 1);
  // pooled
  float* pooled = auf(BATCH * 512);
  _Float16* pooled_h = auh(BATCH * 512);
  float* poolc = auf(BATCH * 512);
  float* poolo = auf(BATCH * 512);
  k_meanpool<<<dim3(512, BATCH), 256, 0, stream>>>(cbuf, pooled, pooled_h, 4096, 512LL * 4096);
  gemmNN(w_pool, pooled_h, poolc, nullptr, nullptr, 512, 1, 512, 512, 1, 1,
         BATCH, 0, 512, 512, 0, 1, 0, 0, 0);
  gnorm(poolc, poolo, nullptr, F(129), F(130), 512, 1, 512, 512, 0, 1);
  k_broadcast<<<cdiv((long long)BATCH * 512 * 4096, 256), 256, 0, stream>>>(
      poolo, catb + 512LL * 4096, 512, 2560LL * 4096, (long long)BATCH * 512 * 4096);

  // window MHA branches
  long long nQ = 8192LL * 512;
  _Float16* qbuf  = auh(nQ);
  _Float16* kvbuf = auh(nQ);
  _Float16 *Qh = auh(nQ), *Kh = auh(nQ), *Vh = auh(nQ), *mh = auh(nQ), *ph = auh(2048LL * 4096);
  float* dumpf  = auf(nQ);
  float* scores = auf(2048LL * 4096);
  float* oproj  = auf(nQ);
  float* ctx    = auf((long long)BATCH * 128 * 4096);
  k_qgather<<<cdiv(nQ, 256), 256, 0, stream>>>(cbuf, qbuf, 512LL * 4096, nQ);
  const int rs[3] = {2, 4, 8};
  for (int j = 0; j < 3; ++j) {
    int r = rs[j], co = 512 / (r * r);
    int di = 142 + 6 * j, ai = 160 + 4 * j;
    k_convr<<<cdiv((long long)BATCH * co * 4096, 256), 256, 0, stream>>>(
        cbuf, F(di), F(di + 1), F(di + 2), F(di + 3), F(di + 4), F(di + 5),
        ctx, co, r, 512LL * 4096, (long long)co * 4096, (long long)BATCH * co * 4096);
    k_kvgather<<<cdiv(nQ, 256), 256, 0, stream>>>(ctx, kvbuf, r, co, (long long)co * 4096, nQ);
    // QKV projections (heads=16, hd=32)
    gemmNT(qbuf,  w_inw[j],               dumpf, Qh, F(ai + 1),        8192, 512, 512, 512, 512, 512, 1, 0, 0, 0, 0, 1, 0, 0, 0);
    gemmNT(kvbuf, w_inw[j] + 512LL * 512, dumpf, Kh, F(ai + 1) + 512,  8192, 512, 512, 512, 512, 512, 1, 0, 0, 0, 0, 1, 0, 0, 0);
    gemmNT(kvbuf, w_inw[j] + 1024LL * 512,dumpf, Vh, F(ai + 1) + 1024, 8192, 512, 512, 512, 512, 512, 1, 0, 0, 0, 0, 1, 0, 0, 0);
    // scores: per (win,head) 64x64 = Q(64x32) K(64x32)^T ; z = win*16+head
    gemmNT(Qh, Kh, scores, nullptr, nullptr, 64, 64, 32, 512, 512, 64,
           2048, 32768, 32768, 65536, 0, 16, 32, 32, 4096);
    k_softmax_row<<<cdiv(2048LL * 64, 256), 256, 0, stream>>>(scores, ph, 64, 0.1767767f, 2048LL * 64);
    // P @ V -> concat heads back into [8192,512]
    gemmNN(ph, Vh, dumpf, mh, nullptr, 64, 32, 64, 64, 512, 512,
           2048, 65536, 32768, 32768, 0, 16, 4096, 32, 32);
    gemmNT(mh, w_outw[j], oproj, nullptr, F(ai + 3), 8192, 512, 512, 512, 512, 512,
           1, 0, 0, 0, 0, 1, 0, 0, 0);
    k_oscatter<<<cdiv((long long)BATCH * 512 * 4096, 256), 256, 0, stream>>>(
        oproj, catb + (1024LL + 512LL * j) * 4096, 2560LL * 4096, (long long)BATCH * 512 * 4096);
  }

  // ---------------- cat conv + low-level fuse + output ------------------------
  long long ncat = (long long)BATCH * 2560 * 4096;
  _Float16* cath = auh(ncat);
  float* catout = auf((long long)BATCH * 512 * 4096);
  k_cvt_h<<<cdiv(ncat, 256), 256, 0, stream>>>(catb, cath, ncat);
  gemmNN(w_cat, cath, catout, nullptr, nullptr, 512, 4096, 2560, 2560, 4096, 4096,
         BATCH, 0, 2560LL * 4096, 512LL * 4096, 0, 1, 0, 0, 0);
  gnorm(catout, catout, nullptr, F(132), F(133), 512, 4096, 512LL * 4096, 512LL * 4096, 0, 1);

  float* lowb = auf((long long)BATCH * 560 * 16384);
  float* lc1o = auf((long long)BATCH * 48 * 16384);
  gemmNN(w_lc1, h_res2, lc1o, nullptr, nullptr, 48, 16384, 256, 256, 16384, 16384,
         BATCH, 0, 256LL * 16384, 48LL * 16384, 0, 1, 0, 0, 0);
  k_bnrelu<<<cdiv((long long)BATCH * 48 * 16384, 256), 256, 0, stream>>>(
      lc1o, lowb + 512LL * 16384, F(138), F(139), F(140), F(141),
      48, 16384, 48LL * 16384, 560LL * 16384, (long long)BATCH * 48 * 16384);
  k_bilinear<<<cdiv((long long)BATCH * 512 * 16384, 256), 256, 0, stream>>>(
      catout, lowb, 512, 64, 128, 512LL * 4096, 560LL * 16384, (long long)BATCH * 512 * 16384);

  long long nlow = (long long)BATCH * 560 * 16384;
  _Float16* lowh = auh(nlow);
  float* fin = auf((long long)BATCH * 256 * 16384);
  k_cvt_h<<<cdiv(nlow, 256), 256, 0, stream>>>(lowb, lowh, nlow);
  gemmNN(w_low, lowh, fin, nullptr, nullptr, 256, 16384, 560, 560, 16384, 16384,
         BATCH, 0, 560LL * 16384, 256LL * 16384, 0, 1, 0, 0, 0);
  gnorm(fin, (float*)d_out, nullptr, F(135), F(136), 256, 16384,
        256LL * 16384, 256LL * 16384, 0, 1);
}
